// P2IKT_model_45535243272823
// MI455X (gfx1250) — matrix operand order repaired and checked by reading, not compiled, not run on hardware
//
#include <hip/hip_runtime.h>
#include <hip/hip_bf16.h>
#include <stdint.h>

// ---------------- CDNA5 WMMA types ----------------
typedef _Float16 v16h __attribute__((ext_vector_type(16)));
typedef _Float16 v8h  __attribute__((ext_vector_type(8)));
typedef _Float16 v4h  __attribute__((ext_vector_type(4)));
typedef float    v8f  __attribute__((ext_vector_type(8)));

#define BB 8
#define HWPIX 65536            // 256*256
#define GROWS 276              // padded rows  (10-halo)
#define GPITCH 280             // padded pitch (10-halo + vector overread slack)

// ---------------- workspace layout ----------------
__host__ __device__ constexpr size_t AL(size_t v) { return (v + 255) & ~size_t(255); }
constexpr size_t SZ_XG   = (size_t)BB * 66 * HWPIX * 2;        // f16 gauss activations
constexpr size_t SZ_XAF  = (size_t)BB * 66 * HWPIX * 2;        // f16 tr activations
constexpr size_t SZ_XPAD = (size_t)BB * 3 * GROWS * GPITCH * 4;// padded f32 input
constexpr size_t SZ_OUT1 = (size_t)BB * HWPIX * 4;
constexpr size_t SZ_PACC = (size_t)BB * 128 * 9 * 4;
constexpr size_t SZ_KERN = (size_t)BB * 1323 * 4;
constexpr size_t SZ_WG   = (size_t)3 * 32 * 512 * 2;           // gauss W, K rows of 24 -> Kpad 512
constexpr size_t SZ_WP1  = (size_t)128 * 416 * 2;              // p1 W,   K rows of 12 -> Kpad 416
constexpr size_t SZ_WFU  = (size_t)80 * 1216 * 2;              // fu W,   Kpad 1216
constexpr size_t SZ_TRS  = (size_t)66 * 9 * 4;

// xg/xaf get 256B guard bands on both sides (branchless +/-1px overreads)
constexpr size_t O_XG   = 0;
constexpr size_t O_XAF  = O_XG   + AL(SZ_XG + 512);
constexpr size_t O_XPAD = O_XAF  + AL(SZ_XAF + 512);
constexpr size_t O_OUT1 = O_XPAD + AL(SZ_XPAD);
constexpr size_t O_PACC = O_OUT1 + AL(SZ_OUT1);
constexpr size_t O_KERN = O_PACC + AL(SZ_PACC);
constexpr size_t O_WG   = O_KERN + AL(SZ_KERN);
constexpr size_t O_WP1  = O_WG   + AL(SZ_WG);
constexpr size_t O_WFU  = O_WP1  + AL(SZ_WP1);
constexpr size_t O_TRS  = O_WFU  + AL(SZ_WFU);

// ---------------- WMMA fragment loaders (documented 16-bit layouts) ----------------
__device__ inline v16h load_a_frag(const _Float16* __restrict__ W, int Kpad, int kc, int lane) {
    int row = lane & 15;
    int kh  = (lane < 16) ? 0 : 8;
    const _Float16* p = W + row * Kpad + kc + kh;
    v8h lo = *(const v8h*)p;            // 16B global load
    v8h hi = *(const v8h*)(p + 16);     // 16B global load
    v16h a;
#pragma unroll
    for (int i = 0; i < 8; ++i) { a[i] = lo[i]; a[i + 8] = hi[i]; }
    return a;
}

// Bs laid out [n][k] (n = pixel 0..15 within tile, k = 0..31)
__device__ inline v16h load_b_frag(const _Float16* Bs, int lane) {
    int n  = lane & 15;
    int kh = (lane < 16) ? 0 : 8;
    const _Float16* p = Bs + n * 32 + kh;
    v8h lo = *(const v8h*)p;            // 16B ds load
    v8h hi = *(const v8h*)(p + 16);     // 16B ds load
    v16h b;
#pragma unroll
    for (int i = 0; i < 8; ++i) { b[i] = lo[i]; b[i + 8] = hi[i]; }
    return b;
}

#define WMMA_F16(a, b, c) __builtin_amdgcn_wmma_f32_16x16x32_f16(false, a, false, b, (short)0, c, false, false)

// ---------------- prep kernels ----------------
__global__ void k_zero_f32(float* p, int n) {
    int i = blockIdx.x * blockDim.x + threadIdx.x;
    if (i < n) p[i] = 0.f;
}

// haloed f32 input: xpad[img][yp][xp], yp/xp offset by 10
__global__ void k_prep_pad(const float* __restrict__ x, float* __restrict__ xpad) {
    int i = blockIdx.x * blockDim.x + threadIdx.x;
    if (i >= 24 * GROWS * GPITCH) return;
    int img = i / (GROWS * GPITCH);
    int r = i - img * (GROWS * GPITCH);
    int yp = r / GPITCH, xp = r - yp * GPITCH;
    int iy = yp - 10, ix = xp - 10;
    float v = 0.f;
    if ((unsigned)iy < 256u && (unsigned)ix < 256u) v = x[((size_t)img << 16) + (iy << 8) + ix];
    xpad[i] = v;
}

// gauss bank -> [3][32][512], K index = ky*24 + kx (kx<21 valid, ky<21 valid)
__global__ void k_prep_gauss(const float* __restrict__ wg, _Float16* __restrict__ Wg) {
    int i = blockIdx.x * blockDim.x + threadIdx.x;
    if (i >= 3 * 32 * 512) return;
    int g = i / (32 * 512);
    int r = i - g * (32 * 512);
    int m = r / 512, kp = r - m * 512;
    int ky = kp / 24, kx = kp - ky * 24;
    float v = 0.f;
    if (m < 22 && ky < 21 && kx < 21) v = wg[(g * 22 + m) * 441 + ky * 21 + kx];
    Wg[i] = (_Float16)v;
}

// p1 -> [128][416], K index = (cin*11+ky)*12 + kx (row<33, kx<11 valid)
__global__ void k_prep_p1(const float* __restrict__ w, _Float16* __restrict__ W) {
    int i = blockIdx.x * blockDim.x + threadIdx.x;
    if (i >= 128 * 416) return;
    int m = i / 416, kp = i - m * 416;
    int r = kp / 12, kx = kp - r * 12;
    float v = 0.f;
    if (r < 33 && kx < 11) v = w[m * 363 + r * 11 + kx];
    W[i] = (_Float16)v;
}

// fu -> [80][1216], K index = cin*9 + ky*3 + kx
__global__ void k_prep_fu(const float* __restrict__ w, _Float16* __restrict__ W) {
    int i = blockIdx.x * blockDim.x + threadIdx.x;
    if (i >= 80 * 1216) return;
    int m = i / 1216, k = i - m * 1216;
    W[i] = (_Float16)((m < 66 && k < 1188) ? w[m * 1188 + k] : 0.f);
}

// tr: fold the 3 identical input channels -> [66][9]
__global__ void k_prep_trsum(const float* __restrict__ w, float* __restrict__ s) {
    int i = blockIdx.x * blockDim.x + threadIdx.x;
    if (i >= 66 * 9) return;
    int m = i / 9, r = i - m * 9;
    s[i] = w[(m * 3 + 0) * 9 + r] + w[(m * 3 + 1) * 9 + r] + w[(m * 3 + 2) * 9 + r];
}

// ---------------- gauss grouped 21x21 conv via WMMA (N=32, double-buffered LDS) ----------------
// grid (2048 ntiles, 3 groups, 8 batch), block 64 (2 waves = 2 M-tiles of 16)
__global__ void k_gauss_wmma(const float* __restrict__ xpad, const _Float16* __restrict__ Wg,
                             _Float16* __restrict__ xg) {
    __shared__ _Float16 Bs[2][32 * 32];
    const int pix0 = blockIdx.x * 32;
    const int g = blockIdx.y, b = blockIdx.z;
    const int y = pix0 >> 8, x0 = pix0 & 255;
    const int tid = threadIdx.x, lane = tid & 31, wave = tid >> 5;
    const float* xin = xpad + (size_t)(b * 3 + g) * (GROWS * GPITCH);
    const _Float16* Wrow = Wg + g * (32 * 512) + wave * 16 * 512;

    auto stage = [&](int kc, _Float16* dst) {
#pragma unroll
        for (int it = 0; it < 4; ++it) {          // 256 quad-tasks / 64 threads
            int t = tid + it * 64;
            int n = t >> 3, q = t & 7;
            int kb = kc + q * 4;
            int ky = kb / 24;
            int kx0 = kb - ky * 24;
            int kyl = ky < 21 ? ky : 20;          // clamp address, mask value
            const float* src = xin + (y + kyl) * GPITCH + (x0 + n + kx0);
            v4h h;
#pragma unroll
            for (int j = 0; j < 4; ++j) {
                float v = src[j];                 // unconditional (halo-padded)
                bool ok = (ky < 21) & (kx0 + j < 21);
                h[j] = ok ? (_Float16)v : (_Float16)0.f;
            }
            *(v4h*)&dst[n * 32 + q * 4] = h;      // one ds_store_b64
        }
    };

    v8f acc0 = {}, acc1 = {};
    stage(0, Bs[0]);
    __syncthreads();
    int pb = 0;
    for (int kc = 0; kc < 512; kc += 32, pb ^= 1) {
        v16h a  = load_a_frag(Wrow, 512, kc, lane);
        v16h b0 = load_b_frag(Bs[pb], lane);
        v16h b1 = load_b_frag(Bs[pb] + 16 * 32, lane);
        if (kc + 32 < 512) stage(kc + 32, Bs[pb ^ 1]);   // overlap with WMMA
        acc0 = WMMA_F16(a, b0, acc0);
        acc1 = WMMA_F16(a, b1, acc1);
        __syncthreads();                          // single barrier per chunk
    }
    int col = lane & 15;
#pragma unroll
    for (int v = 0; v < 8; ++v) {
        int m = wave * 16 + ((lane < 16) ? v : v + 8);
        if (m < 22) {
            int ch = g * 22 + m;
            int base = (((b * 66 + ch) << 8 | y) << 8) + x0 + col;
            xg[base]      = (_Float16)acc0[v];
            xg[base + 16] = (_Float16)acc1[v];
        }
    }
}

// ---------------- p1 11x11 conv (3->128) via WMMA + fused adaptive-pool ----------------
// grid (250 rows * 8 xtiles, 4 ch-blocks, 8 batch), block 64
__global__ void k_p1_wmma(const float* __restrict__ xpad, const _Float16* __restrict__ Wp1,
                          const float* __restrict__ p1b, float* __restrict__ pacc) {
    __shared__ _Float16 Bs[2][32 * 32];
    __shared__ float sacc[32][3];
    const int bx = blockIdx.x;
    const int y = bx >> 3, x0 = (bx & 7) * 32;    // OH=OW=250
    const int chBase = blockIdx.y * 32;
    const int b = blockIdx.z;
    const int tid = threadIdx.x, lane = tid & 31, wave = tid >> 5;
    const _Float16* Wrow = Wp1 + (chBase + wave * 16) * 416;

    auto stage = [&](int kc, _Float16* dst) {
#pragma unroll
        for (int it = 0; it < 4; ++it) {
            int t = tid + it * 64;
            int n = t >> 3, q = t & 7;
            int kb = kc + q * 4;
            int r = kb / 12;
            int kx0 = kb - r * 12;
            int rl = r < 33 ? r : 32;
            int cin = rl / 11, ky = rl - 11 * cin;
            // p1 pad=2 in 10-halo space: row y+ky+8, col x0+n+kx+8
            const float* src = xpad + (size_t)(b * 3 + cin) * (GROWS * GPITCH)
                             + (y + ky + 8) * GPITCH + (x0 + n + kx0 + 8);
            v4h h;
#pragma unroll
            for (int j = 0; j < 4; ++j) {
                float v = src[j];
                bool ok = (r < 33) & (kx0 + j < 11);
                h[j] = ok ? (_Float16)v : (_Float16)0.f;
            }
            *(v4h*)&dst[n * 32 + q * 4] = h;
        }
    };

    v8f acc0 = {}, acc1 = {};
    stage(0, Bs[0]);
    __syncthreads();
    int pb = 0;
    for (int kc = 0; kc < 416; kc += 32, pb ^= 1) {
        v16h a  = load_a_frag(Wrow, 416, kc, lane);
        v16h b0 = load_b_frag(Bs[pb], lane);
        v16h b1 = load_b_frag(Bs[pb] + 16 * 32, lane);
        if (kc + 32 < 416) stage(kc + 32, Bs[pb ^ 1]);
        acc0 = WMMA_F16(a, b0, acc0);
        acc1 = WMMA_F16(a, b1, acc1);
        __syncthreads();
    }
    for (int i = tid; i < 96; i += 64) sacc[i / 3][i % 3] = 0.f;
    __syncthreads();
    int col = lane & 15;
#pragma unroll
    for (int nt = 0; nt < 2; ++nt) {
        int xcol = x0 + nt * 16 + col;
        if (xcol < 250) {
#pragma unroll
            for (int v = 0; v < 8; ++v) {
                int mloc = wave * 16 + ((lane < 16) ? v : v + 8);
                float val = (nt ? acc1[v] : acc0[v]) + p1b[chBase + mloc];
                val = val > 0.f ? val : 0.2f * val;        // leaky before pool
                // bins for L=250,o=3: [0,84),[83,167),[166,250)
                if (xcol < 84) atomicAdd(&sacc[mloc][0], val);
                if (xcol >= 83 && xcol < 167) atomicAdd(&sacc[mloc][1], val);
                if (xcol >= 166) atomicAdd(&sacc[mloc][2], val);
            }
        }
    }
    __syncthreads();
    for (int i = tid; i < 96; i += 64) {
        int mloc = i / 3, wb = i % 3;
        float v = sacc[mloc][wb];
        int base = ((b * 128 + chBase + mloc) * 3) * 3 + wb;
        if (y < 84) atomicAdd(&pacc[base + 0], v);
        if (y >= 83 && y < 167) atomicAdd(&pacc[base + 3], v);
        if (y >= 166) atomicAdd(&pacc[base + 6], v);
    }
}

// ---------------- predictor tail ----------------
__global__ void k_pred_tail(const float* __restrict__ pacc, const float* __restrict__ p2w,
                            const float* __restrict__ p3w, float* __restrict__ kernD) {
    __shared__ float h1[128 * 9];
    __shared__ float h2[441 * 9];
    __shared__ float h3[441 * 3];
    __shared__ float tt[441 * 3];
    __shared__ float red[8];
    const int b = blockIdx.x, tid = threadIdx.x;
    const float invc = 1.f / (84.f * 84.f);
    for (int i = tid; i < 1152; i += 256) h1[i] = pacc[b * 1152 + i] * invc;
    __syncthreads();
    for (int i = tid; i < 3969; i += 256) {
        int o = i / 9, ij = i - o * 9;
        float s = 0.f;
        for (int c = 0; c < 128; ++c) s += p2w[o * 128 + c] * h1[c * 9 + ij];
        h2[i] = s > 0.f ? s : 0.2f * s;
    }
    __syncthreads();
    for (int i = tid; i < 1323; i += 256) {
        int o = i / 3, r = i - o * 3;
        h3[i] = (h2[o * 9 + r * 3] + h2[o * 9 + r * 3 + 1] + h2[o * 9 + r * 3 + 2]) * (1.f / 3.f);
    }
    __syncthreads();
    for (int i = tid; i < 1323; i += 256) {
        int o = i / 3, r = i - o * 3;
        float s = 0.f;
        for (int c = 0; c < 441; ++c) s += p3w[o * 441 + c] * h3[c * 3 + r];
        tt[i] = s;
    }
    __syncthreads();
    if (tid < 3) {
        float mx = -1e30f;
        for (int o = 0; o < 441; ++o) mx = fmaxf(mx, tt[o * 3 + tid]);
        red[tid] = mx;
    }
    __syncthreads();
    for (int i = tid; i < 1323; i += 256) tt[i] = __expf(tt[i] - red[i % 3]);
    __syncthreads();
    if (tid < 3) {
        float s = 0.f;
        for (int o = 0; o < 441; ++o) s += tt[o * 3 + tid];
        red[4 + tid] = 1.f / s;
    }
    __syncthreads();
    for (int i = tid; i < 1323; i += 256) kernD[b * 1323 + i] = tt[i] * red[4 + i % 3];
}

// ---------------- dynamic per-sample 21x21 conv (uses haloed input) ----------------
__global__ void k_dynconv(const float* __restrict__ xpad, const float* __restrict__ kernD,
                          float* __restrict__ out1) {
    __shared__ float kl[1323];
    const int b = blockIdx.z;
    const int tid = threadIdx.x;
    for (int i = tid; i < 1323; i += 256) kl[i] = kernD[b * 1323 + i];
    __syncthreads();
    const int y = blockIdx.y * 16 + (tid >> 4);
    const int xx = blockIdx.x * 16 + (tid & 15);
    float s = 0.f;
    for (int c = 0; c < 3; ++c) {
        const float* xin = xpad + (size_t)(b * 3 + c) * (GROWS * GPITCH) + y * GPITCH + xx;
        const float* kc = kl + c * 441;
        for (int ky = 0; ky < 21; ++ky)
#pragma unroll
            for (int kx = 0; kx < 21; ++kx)
                s += kc[ky * 21 + kx] * xin[ky * GPITCH + kx];
    }
    out1[((size_t)b << 16) + (y << 8) + xx] = s;
}

// ---------------- tr 3x3 conv (channel-folded) ----------------
__global__ void k_tr(const float* __restrict__ out1, const float* __restrict__ trs,
                     const float* __restrict__ trb, _Float16* __restrict__ xaf) {
    int idx = blockIdx.x * blockDim.x + threadIdx.x;
    if (idx >= BB * 66 * HWPIX) return;
    int b = idx / (66 * HWPIX);
    int r = idx - b * (66 * HWPIX);
    int m = r >> 16;
    int p = r & 65535;
    int y = p >> 8, xx = p & 255;
    float s = trb[m];
    const float* in = out1 + ((size_t)b << 16);
#pragma unroll
    for (int ky = 0; ky < 3; ++ky) {
        int iy = y + ky - 1;
        if ((unsigned)iy >= 256u) continue;
#pragma unroll
        for (int kx = 0; kx < 3; ++kx) {
            int ix = xx + kx - 1;
            if ((unsigned)ix < 256u) s += trs[m * 9 + ky * 3 + kx] * in[(iy << 8) + ix];
        }
    }
    xaf[idx] = (_Float16)s;
}

// ---------------- fu 3x3 conv (132 -> 66) via WMMA (N=32, double-buffered LDS) ----------------
// grid 16384 ntiles, block 160 (5 waves = 5 M-tiles over 80 padded channels)
__global__ void k_fu_wmma(const _Float16* __restrict__ xg, const _Float16* __restrict__ xaf,
                          const _Float16* __restrict__ Wfu, const float* __restrict__ fub,
                          float* __restrict__ outF) {
    __shared__ _Float16 Bs[2][32 * 32];
    const int pix0 = blockIdx.x * 32;
    const int b = pix0 >> 16;
    const int rem = pix0 & 65535;
    const int y = rem >> 8, x0 = rem & 255;
    const int tid = threadIdx.x, lane = tid & 31, wave = tid >> 5;
    const _Float16* Wrow = Wfu + wave * 16 * 1216;

    auto stage = [&](int kc, _Float16* dst) {
#pragma unroll
        for (int it = 0; it < 2; ++it) {          // 256 quad-tasks / 160 threads
            int t = tid + it * 160;
            if (t < 256) {
                int n = t >> 3, q = t & 7;
                v4h h;
#pragma unroll
                for (int j = 0; j < 4; ++j) {
                    int kk = kc + q * 4 + j;
                    int c = kk / 9;
                    int r9 = kk - 9 * c;
                    int ky = r9 / 3, kx = r9 - 3 * ky;
                    int iy = y + ky - 1;
                    int iyl = iy < 0 ? 0 : (iy > 255 ? 255 : iy);
                    int ix = x0 + n + kx - 1;     // -1..256: guard bands make it safe
                    int cl = c < 132 ? c : 131;
                    const _Float16* src = (cl < 66)
                        ? xg  + (((b * 66 + cl) << 8 | iyl) << 8) + ix
                        : xaf + (((b * 66 + (cl - 66)) << 8 | iyl) << 8) + ix;
                    _Float16 v = *src;            // unconditional load
                    bool ok = (kk < 1188) & ((unsigned)iy < 256u) & ((unsigned)ix < 256u);
                    h[j] = ok ? v : (_Float16)0.f;
                }
                *(v4h*)&dst[n * 32 + q * 4] = h;
            }
        }
    };

    v8f acc0 = {}, acc1 = {};
    stage(0, Bs[0]);
    __syncthreads();
    int pb = 0;
    for (int kc = 0; kc < 1216; kc += 32, pb ^= 1) {
        if (kc + 32 < 1216) __builtin_prefetch(Wrow + (kc + 32), 0, 3);
        v16h a  = load_a_frag(Wrow, 1216, kc, lane);
        v16h b0 = load_b_frag(Bs[pb], lane);
        v16h b1 = load_b_frag(Bs[pb] + 16 * 32, lane);
        if (kc + 32 < 1216) stage(kc + 32, Bs[pb ^ 1]);   // overlap with WMMA
        acc0 = WMMA_F16(a, b0, acc0);
        acc1 = WMMA_F16(a, b1, acc1);
        __syncthreads();
    }
    int col = lane & 15;
#pragma unroll
    for (int v = 0; v < 8; ++v) {
        int m = wave * 16 + ((lane < 16) ? v : v + 8);
        if (m < 66) {
            int base = (((b * 66 + m) << 8 | y) << 8) + x0 + col;
            outF[base]      = acc0[v] + fub[m];
            outF[base + 16] = acc1[v] + fub[m];
        }
    }
}

// ---------------- ax 3x3 conv (3 -> 3) ----------------
__global__ void k_ax(const float* __restrict__ x, const float* __restrict__ csw,
                     const float* __restrict__ csb, float* __restrict__ outAx) {
    int idx = blockIdx.x * blockDim.x + threadIdx.x;
    if (idx >= BB * 3 * HWPIX) return;
    int b = idx / (3 * HWPIX);
    int r = idx - b * (3 * HWPIX);
    int o = r >> 16;
    int p = r & 65535;
    int y = p >> 8, xx = p & 255;
    float s = csb[o];
    for (int c = 0; c < 3; ++c) {
        const float* in = x + ((size_t)(b * 3 + c) << 16);
#pragma unroll
        for (int ky = 0; ky < 3; ++ky) {
            int iy = y + ky - 1;
            if ((unsigned)iy >= 256u) continue;
#pragma unroll
            for (int kx = 0; kx < 3; ++kx) {
                int ix = xx + kx - 1;
                if ((unsigned)ix < 256u)
                    s += csw[((o * 3 + c) * 3 + ky) * 3 + kx] * in[(iy << 8) + ix];
            }
        }
    }
    outAx[idx] = s;
}

// ---------------- launcher ----------------
extern "C" void kernel_launch(void* const* d_in, const int* in_sizes, int n_in,
                              void* d_out, int out_size, void* d_ws, size_t ws_size,
                              hipStream_t stream) {
    const float* x       = (const float*)d_in[0];
    const float* w_gauss = (const float*)d_in[1];
    const float* p1_w    = (const float*)d_in[2];
    const float* p1_b    = (const float*)d_in[3];
    const float* p2_w    = (const float*)d_in[4];
    const float* p3_w    = (const float*)d_in[5];
    const float* cs_w    = (const float*)d_in[6];
    const float* cs_b    = (const float*)d_in[7];
    const float* tr_w    = (const float*)d_in[8];
    const float* tr_b    = (const float*)d_in[9];
    const float* fu_w    = (const float*)d_in[10];
    const float* fu_b    = (const float*)d_in[11];
    float* outF  = (float*)d_out;
    float* outAx = outF + (size_t)BB * 66 * HWPIX;

    char* ws = (char*)d_ws;
    _Float16* xg   = (_Float16*)(ws + O_XG + 256);    // 256B guard bands
    _Float16* xaf  = (_Float16*)(ws + O_XAF + 256);
    float*    xpad = (float*)(ws + O_XPAD);
    float*    out1 = (float*)(ws + O_OUT1);
    float*    pacc = (float*)(ws + O_PACC);
    float*    kern = (float*)(ws + O_KERN);
    _Float16* Wg   = (_Float16*)(ws + O_WG);
    _Float16* Wp1  = (_Float16*)(ws + O_WP1);
    _Float16* Wfu  = (_Float16*)(ws + O_WFU);
    float*    trs  = (float*)(ws + O_TRS);

    // prep
    k_prep_pad<<<(24 * GROWS * GPITCH + 255) / 256, 256, 0, stream>>>(x, xpad);
    k_prep_gauss<<<(3 * 32 * 512 + 255) / 256, 256, 0, stream>>>(w_gauss, Wg);
    k_prep_p1<<<(128 * 416 + 255) / 256, 256, 0, stream>>>(p1_w, Wp1);
    k_prep_fu<<<(80 * 1216 + 255) / 256, 256, 0, stream>>>(fu_w, Wfu);
    k_prep_trsum<<<(66 * 9 + 255) / 256, 256, 0, stream>>>(tr_w, trs);
    k_zero_f32<<<(BB * 128 * 9 + 255) / 256, 256, 0, stream>>>(pacc, BB * 128 * 9);

    // heavy WMMA convs + predictor + tail
    k_gauss_wmma<<<dim3(2048, 3, BB), 64, 0, stream>>>(xpad, Wg, xg);
    k_p1_wmma<<<dim3(250 * 8, 4, BB), 64, 0, stream>>>(xpad, Wp1, p1_b, pacc);
    k_pred_tail<<<BB, 256, 0, stream>>>(pacc, p2_w, p3_w, kern);
    k_dynconv<<<dim3(16, 16, BB), 256, 0, stream>>>(xpad, kern, out1);
    k_tr<<<(BB * 66 * HWPIX + 255) / 256, 256, 0, stream>>>(out1, trs, tr_b, xaf);
    k_fu_wmma<<<BB * HWPIX / 32, 160, 0, stream>>>(xg, xaf, Wfu, fu_b, outF);
    k_ax<<<(BB * 3 * HWPIX + 255) / 256, 256, 0, stream>>>(x, cs_w, cs_b, outAx);

    (void)in_sizes; (void)n_in; (void)out_size; (void)ws_size;
}
